// FSNRENDORHead_22342419874469
// MI455X (gfx1250) — matrix operand order repaired and checked by reading, not compiled, not run on hardware
//
#include <hip/hip_runtime.h>
#include <hip/hip_bf16.h>
#include <math.h>

// ---------------------------------------------------------------------------
// MI455X (gfx1250) implementation, round 2.
//  * All matmuls / convs / deconvs on v_wmma_f32_16x16x32_bf16.
//  * GEMM stages f32->bf16 tiles through LDS with unconditional float4 global
//    loads (clamped indices) and ds_load_b128 fragment reads.
//  * conv3d/deconv operands pre-packed to bf16 [spatial,Cin] / [tap,Cout,Cin]
//    so the hot loop is pure b128 loads + WMMA (no per-element branches).
// ---------------------------------------------------------------------------

typedef __bf16 bf16_t;
typedef __attribute__((ext_vector_type(16))) __bf16 v16bf;
typedef __attribute__((ext_vector_type(8)))  __bf16 v8bf;
typedef __attribute__((ext_vector_type(8)))  float  v8f;

#define CAT16(LO, HI) __builtin_shufflevector(LO, HI, 0,1,2,3,4,5,6,7,8,9,10,11,12,13,14,15)

enum { ACT_NONE = 0, ACT_RELU = 1, ACT_SIGMOID = 2, ACT_SOFTPLUS = 3 };

static __device__ __forceinline__ float act_apply(float v, int act) {
  switch (act) {
    case ACT_RELU:     return v > 0.f ? v : 0.f;
    case ACT_SIGMOID:  return 1.f / (1.f + __expf(-v));
    case ACT_SOFTPLUS: return v > 20.f ? v : log1pf(__expf(v));
    default:           return v;
  }
}

// ---------------------------------------------------------------------------
// GEMM: C = act(op(A)[M,K] @ op(B)[K,N] + bias).  K must be a multiple of 32
// (true for every call site).  TA: A stored [K,M]; TB: B stored [N,K].
// Block = 128 threads / 4 waves; block tile 64(M) x 16(N), K-step 32.
// OOB handled by index clamping: a clamped source row/col only corrupts
// output rows/cols that the epilogue never stores.
// ---------------------------------------------------------------------------
template <int TA, int TB>
__global__ void gemm_bf16_wmma(const float* __restrict__ A, const float* __restrict__ B,
                               const float* __restrict__ bias, float* __restrict__ C,
                               int M, int N, int K, int outNM, int act,
                               long batchStrideA, long batchStrideC)
{
  __shared__ bf16_t lA[64][32];   // [m][k]
  __shared__ bf16_t lB[16][32];   // [n][k]
  A += (long)blockIdx.z * batchStrideA;
  C += (long)blockIdx.z * batchStrideC;
  const int t    = threadIdx.x;
  const int lane = t & 31;
  const int wave = t >> 5;
  const int m0 = blockIdx.y * 64;
  const int n0 = blockIdx.x * 16;
  const int baseA = (lane >= 16) ? 8 : 0;    // ISA A-frag k chunks {b..b+7, b+16..b+23}
  const int baseB = (lane >= 16) ? 8 : 0;    // ISA B-frag k chunks {b..b+7, b+16..b+23}
  const int mrow  = wave * 16 + (lane & 15);

  v8f acc = {0.f, 0.f, 0.f, 0.f, 0.f, 0.f, 0.f, 0.f};
  for (int k0 = 0; k0 < K; k0 += 32) {
    // ---- cooperative A slab: 64 x 32 ----
    if (TA == 0) {                                  // A [M,K], k contiguous
      const int am = t >> 1, ak = (t & 1) * 16;
      const long gm = (m0 + am < M) ? (long)(m0 + am) : (long)(M - 1);
      const float4* src = (const float4*)(A + gm * K + k0 + ak);
      if (k0 + 32 < K) __builtin_prefetch(A + gm * K + (k0 + 32) + ak, 0, 1);
#pragma unroll
      for (int i = 0; i < 4; ++i) {
        const float4 v = src[i];
        lA[am][ak + 4 * i + 0] = (bf16_t)v.x;
        lA[am][ak + 4 * i + 1] = (bf16_t)v.y;
        lA[am][ak + 4 * i + 2] = (bf16_t)v.z;
        lA[am][ak + 4 * i + 3] = (bf16_t)v.w;
      }
    } else {                                        // A [K,M], m contiguous
      const int ak = t >> 2, am = (t & 3) * 16;
      const long rb = (long)(k0 + ak) * M;
      if ((m0 + am + 15) < M) {
#pragma unroll
        for (int i = 0; i < 16; ++i)
          lA[am + i][ak] = (bf16_t)A[rb + m0 + am + i];
      } else {
#pragma unroll
        for (int i = 0; i < 16; ++i) {
          int mm = m0 + am + i; if (mm >= M) mm = M - 1;
          lA[am + i][ak] = (bf16_t)A[rb + mm];
        }
      }
    }
    // ---- cooperative B slab: 32 x 16 (stored [n][k]) ----
    if (TB == 1) {                                  // B [N,K], k contiguous
      const int bn = t >> 3, bk = (t & 7) * 4;
      const long gn = (n0 + bn < N) ? (long)(n0 + bn) : (long)(N - 1);
      const float4 v = *(const float4*)(B + gn * K + k0 + bk);
      lB[bn][bk + 0] = (bf16_t)v.x;
      lB[bn][bk + 1] = (bf16_t)v.y;
      lB[bn][bk + 2] = (bf16_t)v.z;
      lB[bn][bk + 3] = (bf16_t)v.w;
    } else {                                        // B [K,N], n contiguous
      const int bk = t >> 2, bn = (t & 3) * 4;
      const long rb = (long)(k0 + bk) * N;
#pragma unroll
      for (int i = 0; i < 4; ++i) {
        int nn = n0 + bn + i; if (nn >= N) nn = N - 1;
        lB[bn + i][bk] = (bf16_t)B[rb + nn];
      }
    }
    __syncthreads();
    // ---- fragments from LDS (ds_load_b128) + WMMA ----
    const v8bf alo = *(const v8bf*)&lA[mrow][baseA];
    const v8bf ahi = *(const v8bf*)&lA[mrow][baseA + 16];
    const v8bf blo = *(const v8bf*)&lB[lane & 15][baseB];
    const v8bf bhi = *(const v8bf*)&lB[lane & 15][baseB + 16];
    const v16bf af  = CAT16(alo, ahi);
    const v16bf bfr = CAT16(blo, bhi);
    acc = __builtin_amdgcn_wmma_f32_16x16x32_bf16(false, af, false, bfr,
                                                  (short)0, acc, false, false);
    __syncthreads();
  }
  const int nOut = n0 + (lane & 15);
  if (nOut >= N) return;
  const float bz = bias ? bias[nOut] : 0.f;
  const int mBase = m0 + wave * 16 + ((lane >= 16) ? 8 : 0);
#pragma unroll
  for (int r = 0; r < 8; ++r) {
    const int mr = mBase + r;
    if (mr < M) {
      const float v = act_apply(acc[r] + bz, act);
      if (outNM) C[(long)nOut * M + mr] = v;
      else       C[(long)mr * N + nOut] = v;
    }
  }
}

// ---------------------------------------------------------------------------
// bf16 pre-packing kernels (run per launch; deterministic).
// ---------------------------------------------------------------------------
// X [C, S] f32 -> Xp [S, C] bf16
__global__ void pack_xT_bf16(const float* __restrict__ X, bf16_t* __restrict__ Xp,
                             int C, long S)
{
  const long i = blockIdx.x * (long)blockDim.x + threadIdx.x;
  if (i >= (long)C * S) return;
  const long s = i / C;
  const int  c = (int)(i % C);
  Xp[i] = (bf16_t)X[(long)c * S + s];
}
// conv weight [Cout,Cin,27] f32 -> [27,Cout,Cin] bf16
__global__ void pack_convw_bf16(const float* __restrict__ Wt, bf16_t* __restrict__ Wp,
                                int Cin, int Cout)
{
  const long i = blockIdx.x * (long)blockDim.x + threadIdx.x;
  if (i >= (long)27 * Cout * Cin) return;
  const int  ci = (int)(i % Cin);
  const long r  = i / Cin;
  const int  co = (int)(r % Cout);
  const int  tp = (int)(r / Cout);
  Wp[i] = (bf16_t)Wt[((long)co * Cin + ci) * 27 + tp];
}
// deconv weight [Cin,Cout,2,2,2] f32 -> [8,Cout,Cin] bf16
__global__ void pack_deconvw_bf16(const float* __restrict__ Wt, bf16_t* __restrict__ Wp,
                                  int Cin, int Cout)
{
  const long i = blockIdx.x * (long)blockDim.x + threadIdx.x;
  if (i >= (long)8 * Cout * Cin) return;
  const int  ci = (int)(i % Cin);
  const long r  = i / Cin;
  const int  co = (int)(r % Cout);
  const int  tp = (int)(r / Cout);
  Wp[i] = (bf16_t)Wt[((long)ci * Cout + co) * 8 + tp];
}

// ---------------------------------------------------------------------------
// Implicit-GEMM 3x3x3 conv, pad 1, on packed bf16 operands.
// Xp:[S,Cin] bf16  Wp:[27,Cout,Cin] bf16  Y:[Cout,W,H,Z] f32
// ---------------------------------------------------------------------------
__global__ void conv3d_k3_bf16(const bf16_t* __restrict__ Xp, const bf16_t* __restrict__ Wp,
                               const float* __restrict__ bias, float* __restrict__ Y,
                               int Cin, int Cout, int Wd, int Hd, int Zd, int doRelu)
{
  const int M = Wd * Hd * Zd;
  const int lane = threadIdx.x & 31;
  const int wave = threadIdx.x >> 5;
  const int row0 = (blockIdx.y * 4 + wave) * 16;
  const int col0 = blockIdx.x * 16;
  const int mA  = row0 + (lane & 15);
  const bool mOk = mA < M;
  const int mAc = mOk ? mA : 0;
  const int aw_ = mAc / (Hd * Zd);
  const int ah  = (mAc / Zd) % Hd;
  const int az  = mAc % Zd;
  const int nB  = col0 + (lane & 15);
  const long nBc = (nB < Cout) ? nB : (Cout - 1);
  const int baseA = (lane >= 16) ? 8 : 0;
  const int baseB = (lane >= 16) ? 8 : 0;

  v8f acc = {0.f, 0.f, 0.f, 0.f, 0.f, 0.f, 0.f, 0.f};
  for (int t = 0; t < 27; ++t) {
    const int dw = t / 9 - 1, dh = (t / 3) % 3 - 1, dz = t % 3 - 1;
    const int sw = aw_ + dw, sh = ah + dh, sz = az + dz;
    const bool sOk = mOk && (unsigned)sw < (unsigned)Wd && (unsigned)sh < (unsigned)Hd
                          && (unsigned)sz < (unsigned)Zd;
    const bf16_t* arow = Xp + (sOk ? (((long)sw * Hd + sh) * Zd + sz) * Cin : 0);
    const bf16_t* brow = Wp + ((long)t * Cout + nBc) * Cin;
    for (int k0 = 0; k0 < Cin; k0 += 32) {
      v8bf alo = {}, ahi = {};
      if (sOk) {                                    // boundary taps must be zero
        alo = *(const v8bf*)(arow + k0 + baseA);
        ahi = *(const v8bf*)(arow + k0 + baseA + 16);
      }
      const v8bf blo = *(const v8bf*)(brow + k0 + baseB);
      const v8bf bhi = *(const v8bf*)(brow + k0 + baseB + 16);
      const v16bf af  = CAT16(alo, ahi);
      const v16bf bfr = CAT16(blo, bhi);
      acc = __builtin_amdgcn_wmma_f32_16x16x32_bf16(false, af, false, bfr,
                                                    (short)0, acc, false, false);
    }
  }
  const int nOut = col0 + (lane & 15);
  if (nOut >= Cout) return;
  const float bz = bias ? bias[nOut] : 0.f;
  const int mBase = row0 + ((lane >= 16) ? 8 : 0);
#pragma unroll
  for (int r = 0; r < 8; ++r) {
    const int mr = mBase + r;
    if (mr < M) {
      float v = acc[r] + bz;
      if (doRelu) v = v > 0.f ? v : 0.f;
      Y[(long)nOut * M + mr] = v;
    }
  }
}

// ---------------------------------------------------------------------------
// ConvTranspose3d, kernel==stride==2, pad 0 (no tap overlap), packed bf16.
//   Y[co, 2w+di, 2h+dj, 2z+dk] = sum_ci X[ci,w,h,z] * W[ci,co,di,dj,dk]
// Xp:[M,Cin] bf16  Wp:[8,Cout,Cin] bf16.  blockIdx.z = tap.
// ---------------------------------------------------------------------------
__global__ void deconv2_bf16(const bf16_t* __restrict__ Xp, const bf16_t* __restrict__ Wp,
                             float* __restrict__ Y, int Cin, int Cout,
                             int Wd, int Hd, int Zd)
{
  const int M = Wd * Hd * Zd;
  const int tap = blockIdx.z;
  const int di = (tap >> 2) & 1, dj = (tap >> 1) & 1, dk = tap & 1;
  const int lane = threadIdx.x & 31;
  const int wave = threadIdx.x >> 5;
  const int row0 = (blockIdx.y * 4 + wave) * 16;
  const int col0 = blockIdx.x * 16;
  const int mA  = row0 + (lane & 15);
  const long mAc = (mA < M) ? mA : (M - 1);
  const int nB  = col0 + (lane & 15);
  const long nBc = (nB < Cout) ? nB : (Cout - 1);
  const int baseA = (lane >= 16) ? 8 : 0;
  const int baseB = (lane >= 16) ? 8 : 0;
  const bf16_t* arow = Xp + mAc * Cin;
  const bf16_t* brow = Wp + ((long)tap * Cout + nBc) * Cin;

  v8f acc = {0.f, 0.f, 0.f, 0.f, 0.f, 0.f, 0.f, 0.f};
  for (int k0 = 0; k0 < Cin; k0 += 32) {
    const v8bf alo = *(const v8bf*)(arow + k0 + baseA);
    const v8bf ahi = *(const v8bf*)(arow + k0 + baseA + 16);
    const v8bf blo = *(const v8bf*)(brow + k0 + baseB);
    const v8bf bhi = *(const v8bf*)(brow + k0 + baseB + 16);
    const v16bf af  = CAT16(alo, ahi);
    const v16bf bfr = CAT16(blo, bhi);
    acc = __builtin_amdgcn_wmma_f32_16x16x32_bf16(false, af, false, bfr,
                                                  (short)0, acc, false, false);
  }
  const int nOut = col0 + (lane & 15);
  if (nOut >= Cout) return;
  const int mBase = row0 + ((lane >= 16) ? 8 : 0);
#pragma unroll
  for (int r = 0; r < 8; ++r) {
    const int mr = mBase + r;
    if (mr < M) {
      const int w = mr / (Hd * Zd);
      const int h = (mr / Zd) % Hd;
      const int z = mr % Zd;
      const long oi = ((((long)nOut * (2 * Wd) + (2 * w + di)) * (2 * Hd)) + (2 * h + dj))
                      * (2 * Zd) + (2 * z + dk);
      Y[oi] = acc[r];
    }
  }
}

// ------------------------- scalar support kernels --------------------------

__global__ void ln_residual(const float* __restrict__ A, const float* __restrict__ Bv,
                            const float* __restrict__ G, const float* __restrict__ Be,
                            float* __restrict__ O, int C)
{
  const int lane = threadIdx.x;        // blockDim = 32 (wave32)
  const long base = (long)blockIdx.x * C;
  float s = 0.f, s2 = 0.f;
  for (int i = lane; i < C; i += 32) {
    const float v = A[base + i] + Bv[base + i];
    s += v; s2 += v * v;
  }
  for (int off = 16; off; off >>= 1) {
    s  += __shfl_xor(s,  off, 32);
    s2 += __shfl_xor(s2, off, 32);
  }
  const float mean = s / C;
  const float var  = s2 / C - mean * mean;
  const float rstd = rsqrtf(var + 1e-5f);
  for (int i = lane; i < C; i += 32) {
    const float v = A[base + i] + Bv[base + i];
    O[base + i] = (v - mean) * rstd * G[i] + Be[i];
  }
}

__global__ void gn_stats(const float* __restrict__ X, float* __restrict__ stats,
                         int C, long S, int groups)
{
  __shared__ float sh[256], sh2[256];
  const int g = blockIdx.x;
  const long n = (long)(C / groups) * S;
  const long base = (long)g * n;
  float s = 0.f, s2 = 0.f;
  for (long i = threadIdx.x; i < n; i += blockDim.x) {
    const float v = X[base + i]; s += v; s2 += v * v;
  }
  sh[threadIdx.x] = s; sh2[threadIdx.x] = s2;
  __syncthreads();
  for (int off = blockDim.x / 2; off; off >>= 1) {
    if ((int)threadIdx.x < off) {
      sh[threadIdx.x]  += sh[threadIdx.x + off];
      sh2[threadIdx.x] += sh2[threadIdx.x + off];
    }
    __syncthreads();
  }
  if (threadIdx.x == 0) {
    const float mean = sh[0] / n;
    const float var  = sh2[0] / n - mean * mean;
    stats[2 * g]     = mean;
    stats[2 * g + 1] = rsqrtf(var + 1e-5f);
  }
}

__global__ void gn_apply_relu(float* __restrict__ X, const float* __restrict__ stats,
                              const float* __restrict__ G, const float* __restrict__ B,
                              int C, long S, int groups)
{
  const long i = blockIdx.x * (long)blockDim.x + threadIdx.x;
  if (i >= (long)C * S) return;
  const int c = (int)(i / S);
  const int g = c / (C / groups);
  const float v = (X[i] - stats[2 * g]) * stats[2 * g + 1] * G[c] + B[c];
  X[i] = v > 0.f ? v : 0.f;
}

__global__ void add_inplace_k(float* __restrict__ Y, const float* __restrict__ Xs, long n)
{
  const long i = blockIdx.x * (long)blockDim.x + threadIdx.x;
  if (i < n) Y[i] += Xs[i];
}

__global__ void softmax_lastdim(float* __restrict__ X, long rows, int P)
{
  const long r = blockIdx.x * (long)blockDim.x + threadIdx.x;
  if (r >= rows) return;
  float* x = X + r * P;
  float mx = x[0];
  for (int p = 1; p < P; ++p) mx = fmaxf(mx, x[p]);
  float s = 0.f;
  for (int p = 0; p < P; ++p) { const float e = __expf(x[p] - mx); x[p] = e; s += e; }
  const float inv = 1.f / s;
  for (int p = 0; p < P; ++p) x[p] *= inv;
}

// Deformable bilinear sampling + attention weights + camera mean.
// V:[cams,H*W,C]  REF:[Nq,2]  OFF:[Nq,Nh,P,2]  AW:[Nq,Nh,P]  OUT:[Nq,C]
__global__ void deform_sample(const float* __restrict__ V, const float* __restrict__ REF,
                              const float* __restrict__ OFF, const float* __restrict__ AW,
                              float* __restrict__ OUT,
                              int Nq, int C, int Nh, int P, int H, int W, int cams)
{
  const long tid = blockIdx.x * (long)blockDim.x + threadIdx.x;
  if (tid >= (long)Nq * C) return;
  const int q = (int)(tid / C), c = (int)(tid % C);
  const int hd = C / Nh, h = c / hd;
  const long HWl = (long)H * W;
  const float rx = REF[2 * q], ry = REF[2 * q + 1];
  float acc = 0.f;
  for (int p = 0; p < P; ++p) {
    const long ob = (((long)q * Nh + h) * P + p) * 2;
    const float lx = rx + OFF[ob]     / (float)W;
    const float ly = ry + OFF[ob + 1] / (float)H;
    const float x = lx * W - 0.5f, y = ly * H - 0.5f;
    const float x0f = floorf(x), y0f = floorf(y);
    const float dx = x - x0f, dy = y - y0f;
    const int x0 = (int)x0f, y0 = (int)y0f;
    const float wgt = AW[((long)q * Nh + h) * P + p];
    for (int cam = 0; cam < cams; ++cam) {
      const float* vb = V + (long)cam * HWl * C + c;
      float sv = 0.f;
      if (x0 >= 0 && x0 < W && y0 >= 0 && y0 < H)
        sv += vb[((long)y0 * W + x0) * C] * (1.f - dx) * (1.f - dy);
      if (x0 + 1 >= 0 && x0 + 1 < W && y0 >= 0 && y0 < H)
        sv += vb[((long)y0 * W + x0 + 1) * C] * dx * (1.f - dy);
      if (x0 >= 0 && x0 < W && y0 + 1 >= 0 && y0 + 1 < H)
        sv += vb[((long)(y0 + 1) * W + x0) * C] * (1.f - dx) * dy;
      if (x0 + 1 >= 0 && x0 + 1 < W && y0 + 1 >= 0 && y0 + 1 < H)
        sv += vb[((long)(y0 + 1) * W + x0 + 1) * C] * dx * dy;
      acc += wgt * sv;
    }
  }
  OUT[(long)q * C + c] = acc / (float)cams;
}

// ve[Nq,C] (q=(z*VH+h)*VW+w) -> vol[C,VW,VH,VZ]
__global__ void ve_to_vol(const float* __restrict__ VE, float* __restrict__ VOL,
                          int C, int VWd, int VHd, int VZd)
{
  const long tid = blockIdx.x * (long)blockDim.x + threadIdx.x;
  const long n = (long)VWd * VHd * VZd * C;
  if (tid >= n) return;
  const int c = (int)(tid % C);
  const long q = tid / C;
  const int w = (int)(q % VWd);
  const int h = (int)((q / VWd) % VHd);
  const int z = (int)(q / ((long)VWd * VHd));
  VOL[(((long)c * VWd + w) * VHd + h) * VZd + z] = VE[q * C + c];
}

// vx[C,W,H,Z] -> out[(z*H+h)*W+w, C]
__global__ void chw_to_hwc(const float* __restrict__ VX, float* __restrict__ OUT,
                           int Cc, int Wd, int Hd, int Zd)
{
  const long tid = blockIdx.x * (long)blockDim.x + threadIdx.x;
  const long n = (long)Cc * Wd * Hd * Zd;
  if (tid >= n) return;
  const int c = (int)(tid % Cc);
  const long m = tid / Cc;
  const int w = (int)(m % Wd);
  const int h = (int)((m / Wd) % Hd);
  const int z = (int)(m / ((long)Wd * Hd));
  OUT[tid] = VX[(((long)c * Wd + w) * Hd + h) * Zd + z];
}

__global__ void stride_extract(const float* __restrict__ IN, float* __restrict__ OUT,
                               long n, int stride, int off)
{
  const long i = blockIdx.x * (long)blockDim.x + threadIdx.x;
  if (i < n) OUT[i] = IN[i * stride + off];
}

// ---------------------------------------------------------------------------
// Host orchestration
// ---------------------------------------------------------------------------

// Input flattening assumption: jax pytree order (dict keys sorted).
enum {
  IN_FEAT0 = 0, IN_FEAT1 = 1, IN_FEAT2 = 2,
  IN_DB0 = 3,                    // per deblock i: +3*i: [gn_b, gn_g, w]
  IN_D1B = 24, IN_D1W = 25, IN_D2B = 26, IN_D2W = 27,
  IN_FINAL_B = 28, IN_FINAL_W = 29,
  IN_LVL0 = 30,                  // 21 entries per level
  IN_OCC0 = 93,
  IN_S1B = 97, IN_S1W = 98, IN_S2B = 99, IN_S2W = 100
};
enum {
  L_AWB = 0, L_AWW, L_EMB, L_F1B, L_F1W, L_F2B, L_F2W,
  L_LN1B, L_LN1G, L_LN2B, L_LN2G, L_OB, L_OW,
  L_OFFB, L_OFFW, L_REFB, L_REFW, L_TB, L_TW, L_VB, L_VW
};

// workspace arena (float units)
static constexpr long OFF_VOL0  = 0;                       // 128*80000
static constexpr long OFF_VOL1  = OFF_VOL0 + 10240000;     // 256*10000
static constexpr long OFF_VOL2  = OFF_VOL1 + 2560000;      // 512*1250
static constexpr long OFF_T0    = OFF_VOL2 + 640000;       // 41M ping
static constexpr long OFF_T1    = OFF_T0 + 41000000;       // 41M pong
static constexpr long OFF_T2    = OFF_T1 + 41000000;       // 12M temp
static constexpr long OFF_SMALL = OFF_T2 + 12000000;       // 10M small
static constexpr long OFF_PKA   = OFF_SMALL + 10000000;    // 21M f32 = 42M bf16
static constexpr long OFF_PKW   = OFF_PKA + 21000000;      // 2M f32 = 4M bf16

// output offsets (floats) in reference return order
static constexpr long OUT_OCC0 = 0;
static constexpr long OUT_OCC1 = 21250;
static constexpr long OUT_OCC2 = 191250;
static constexpr long OUT_OCC3 = 1551250;
static constexpr long OUT_DENS = 12431250;
static constexpr long OUT_SEM  = 12511250;

static void launch_gemm(hipStream_t st, const float* A, const float* B, const float* bias,
                        float* C, int M, int N, int K, int tA, int tB, int outNM, int act,
                        int batch, long sA, long sC)
{
  dim3 g((N + 15) / 16, (M + 63) / 64, batch);
  if (tA && tB)
    gemm_bf16_wmma<1, 1><<<g, 128, 0, st>>>(A, B, bias, C, M, N, K, outNM, act, sA, sC);
  else
    gemm_bf16_wmma<0, 0><<<g, 128, 0, st>>>(A, B, bias, C, M, N, K, outNM, act, sA, sC);
}

extern "C" void kernel_launch(void* const* d_in, const int* in_sizes, int n_in,
                              void* d_out, int out_size, void* d_ws, size_t ws_size,
                              hipStream_t stream)
{
  (void)in_sizes; (void)n_in; (void)out_size; (void)ws_size;
  auto F = [&](int i) { return (const float*)d_in[i]; };
  float* ws  = (float*)d_ws;
  float* out = (float*)d_out;

  float* vol[3] = { ws + OFF_VOL0, ws + OFF_VOL1, ws + OFF_VOL2 };
  float* t0 = ws + OFF_T0;
  float* t1 = ws + OFF_T1;
  float* t2 = ws + OFF_T2;
  float* refb  = ws + OFF_SMALL;
  float* offb  = ws + OFF_SMALL + 200000;
  float* awb   = ws + OFF_SMALL + 2800000;
  float* gns   = ws + OFF_SMALL + 4200000;
  float* dprob = ws + OFF_SMALL + 4300000;
  bf16_t* pkA = (bf16_t*)(ws + OFF_PKA);
  bf16_t* pkW = (bf16_t*)(ws + OFF_PKW);

  auto run_conv = [&](const float* X, const float* Wt, const float* bias, float* Y,
                      int Cin, int Cout, int Wd, int Hd, int Zd, int relu) {
    const long S = (long)Wd * Hd * Zd;
    const long nx = (long)Cin * S;
    pack_xT_bf16<<<(unsigned)((nx + 255) / 256), 256, 0, stream>>>(X, pkA, Cin, S);
    const long nw = 27L * Cout * Cin;
    pack_convw_bf16<<<(unsigned)((nw + 255) / 256), 256, 0, stream>>>(Wt, pkW, Cin, Cout);
    dim3 g((Cout + 15) / 16, (unsigned)((S + 63) / 64));
    conv3d_k3_bf16<<<g, 128, 0, stream>>>(pkA, pkW, bias, Y, Cin, Cout, Wd, Hd, Zd, relu);
  };
  auto run_deconv = [&](const float* X, const float* Wt, float* Y,
                        int Cin, int Cout, int Wd, int Hd, int Zd) {
    const long S = (long)Wd * Hd * Zd;
    const long nx = (long)Cin * S;
    pack_xT_bf16<<<(unsigned)((nx + 255) / 256), 256, 0, stream>>>(X, pkA, Cin, S);
    const long nw = 8L * Cout * Cin;
    pack_deconvw_bf16<<<(unsigned)((nw + 255) / 256), 256, 0, stream>>>(Wt, pkW, Cin, Cout);
    dim3 g((Cout + 15) / 16, (unsigned)((S + 63) / 64), 8);
    deconv2_bf16<<<g, 128, 0, stream>>>(pkA, pkW, Y, Cin, Cout, Wd, Hd, Zd);
  };
  auto run_gn = [&](float* X, const float* g_, const float* b_, int C, long S) {
    gn_stats<<<16, 256, 0, stream>>>(X, gns, C, S, 16);
    const long n = (long)C * S;
    gn_apply_relu<<<(unsigned)((n + 255) / 256), 256, 0, stream>>>(X, gns, g_, b_, C, S, 16);
  };

  static const int EMBEDc[3] = {128, 256, 512};
  static const int VHc[3] = {100, 50, 25}, VWc[3] = {100, 50, 25}, VZc[3] = {8, 4, 2};
  static const int FH[3] = {58, 29, 15}, FW[3] = {100, 50, 25}, NP[3] = {2, 4, 8};

  // ---------------- deformable attention levels -> volumes ----------------
  for (int i = 0; i < 3; ++i) {
    const int C = EMBEDc[i], P = NP[i], H = FH[i], Wf = FW[i], HW = H * Wf;
    const int Nq = VHc[i] * VWc[i] * VZc[i];
    const int L = IN_LVL0 + 21 * i;
    const float* emb = F(L + L_EMB);

    // transfer 1x1 conv (+ReLU), batched over 6 cams: feat [K=512, HW] col-major
    launch_gemm(stream, F(i), F(L + L_TW), F(L + L_TB), t0, HW, C, 512,
                1, 1, 0, ACT_RELU, 6, (long)512 * HW, (long)HW * C);
    // value projection -> vmat [cam, HW, C]
    launch_gemm(stream, t0, F(L + L_VW), F(L + L_VB), t1, HW, C, C,
                0, 0, 0, ACT_NONE, 6, (long)HW * C, (long)HW * C);
    // query heads
    launch_gemm(stream, emb, F(L + L_REFW), F(L + L_REFB), refb, Nq, 2, C,
                0, 0, 0, ACT_SIGMOID, 1, 0, 0);
    launch_gemm(stream, emb, F(L + L_OFFW), F(L + L_OFFB), offb, Nq, 8 * P * 2, C,
                0, 0, 0, ACT_NONE, 1, 0, 0);
    launch_gemm(stream, emb, F(L + L_AWW), F(L + L_AWB), awb, Nq, 8 * P, C,
                0, 0, 0, ACT_NONE, 1, 0, 0);
    softmax_lastdim<<<(unsigned)(((long)Nq * 8 + 255) / 256), 256, 0, stream>>>(awb, (long)Nq * 8, P);
    // sampling + weighted sum + camera mean -> out_pre [Nq, C]
    const long nt = (long)Nq * C;
    deform_sample<<<(unsigned)((nt + 255) / 256), 256, 0, stream>>>(
        t1, refb, offb, awb, t2, Nq, C, 8, P, H, Wf, 6);
    // output projection -> attn_out (t0)
    launch_gemm(stream, t2, F(L + L_OW), F(L + L_OB), t0, Nq, C, C, 0, 0, 0, ACT_NONE, 1, 0, 0);
    // x = LN(emb + attn_out) -> t1
    ln_residual<<<Nq, 32, 0, stream>>>(emb, t0, F(L + L_LN1G), F(L + L_LN1B), t1, C);
    // FFN
    launch_gemm(stream, t1, F(L + L_F1W), F(L + L_F1B), t0, Nq, 2 * C, C,
                0, 0, 0, ACT_RELU, 1, 0, 0);
    launch_gemm(stream, t0, F(L + L_F2W), F(L + L_F2B), t2, Nq, C, 2 * C,
                0, 0, 0, ACT_NONE, 1, 0, 0);
    // ve = LN(x + h) -> t0, then permute to volume [C, VW, VH, VZ]
    ln_residual<<<Nq, 32, 0, stream>>>(t1, t2, F(L + L_LN2G), F(L + L_LN2B), t0, C);
    ve_to_vol<<<(unsigned)((nt + 255) / 256), 256, 0, stream>>>(t0, vol[i], C, VWc[i], VHc[i], VZc[i]);
  }

  // ---------------- density / semantic heads on finest volume -------------
  run_conv(vol[0], F(IN_FINAL_W), F(IN_FINAL_B), t0, 128, 32, 100, 100, 8, 1);
  chw_to_hwc<<<(unsigned)((2560000 + 255) / 256), 256, 0, stream>>>(t0, t1, 32, 100, 100, 8);
  launch_gemm(stream, t1, F(IN_D1W), F(IN_D1B), t2, 80000, 64, 32, 0, 0, 0, ACT_SOFTPLUS, 1, 0, 0);
  launch_gemm(stream, t2, F(IN_D2W), F(IN_D2B), dprob, 80000, 2, 64, 0, 0, 0, ACT_NONE, 1, 0, 0);
  stride_extract<<<(unsigned)((80000 + 255) / 256), 256, 0, stream>>>(dprob, out + OUT_DENS, 80000, 2, 0);
  launch_gemm(stream, t1, F(IN_S1W), F(IN_S1B), t2, 80000, 64, 32, 0, 0, 0, ACT_SOFTPLUS, 1, 0, 0);
  launch_gemm(stream, t2, F(IN_S2W), F(IN_S2B), out + OUT_SEM, 80000, 16, 64, 0, 0, 0, ACT_NONE, 1, 0, 0);

  // ---------------- decoder (ping-pong t0/t1) ------------------------------
  auto dbW = [&](int i) { return F(IN_DB0 + 3 * i + 2); };
  auto dbG = [&](int i) { return F(IN_DB0 + 3 * i + 1); };
  auto dbB = [&](int i) { return F(IN_DB0 + 3 * i + 0); };

  run_conv(vol[2], dbW(0), nullptr, t0, 512, 256, 25, 25, 2, 0);
  run_gn(t0, dbG(0), dbB(0), 256, 1250);
  launch_gemm(stream, t0, F(IN_OCC0 + 0), nullptr, out + OUT_OCC0, 1250, 17, 256, 1, 1, 1, ACT_NONE, 1, 0, 0);

  run_deconv(t0, dbW(1), t1, 256, 256, 25, 25, 2);
  run_gn(t1, dbG(1), dbB(1), 256, 10000);
  add_inplace_k<<<(unsigned)((2560000 + 255) / 256), 256, 0, stream>>>(t1, vol[1], 2560000);

  run_conv(t1, dbW(2), nullptr, t0, 256, 128, 50, 50, 4, 0);
  run_gn(t0, dbG(2), dbB(2), 128, 10000);
  launch_gemm(stream, t0, F(IN_OCC0 + 1), nullptr, out + OUT_OCC1, 10000, 17, 128, 1, 1, 1, ACT_NONE, 1, 0, 0);

  run_deconv(t0, dbW(3), t1, 128, 128, 50, 50, 4);
  run_gn(t1, dbG(3), dbB(3), 128, 80000);
  add_inplace_k<<<(unsigned)((10240000 + 255) / 256), 256, 0, stream>>>(t1, vol[0], 10240000);

  run_conv(t1, dbW(4), nullptr, t0, 128, 64, 100, 100, 8, 0);
  run_gn(t0, dbG(4), dbB(4), 64, 80000);
  launch_gemm(stream, t0, F(IN_OCC0 + 2), nullptr, out + OUT_OCC2, 80000, 17, 64, 1, 1, 1, ACT_NONE, 1, 0, 0);

  run_deconv(t0, dbW(5), t1, 64, 64, 100, 100, 8);
  run_gn(t1, dbG(5), dbB(5), 64, 640000);

  run_conv(t1, dbW(6), nullptr, t0, 64, 32, 200, 200, 16, 0);
  run_gn(t0, dbG(6), dbB(6), 32, 640000);
  launch_gemm(stream, t0, F(IN_OCC0 + 3), nullptr, out + OUT_OCC3, 640000, 17, 32, 1, 1, 1, ACT_NONE, 1, 0, 0);
}
